// GRUPositionEncDec_17111149707909
// MI455X (gfx1250) — matrix operand order repaired and checked by reading, not compiled, not run on hardware
//
#include <hip/hip_runtime.h>

// ---------------------------------------------------------------------------
// GRU encoder-decoder, fused persistent kernel for gfx1250 (MI455X).
// 256 blocks x 128 threads (4 wave32 waves); each block owns 16 batch rows.
// Wave w owns gate tiles {r,z,n} for hidden slice k in [16w, 16w+16):
//   - recurrent GEMM via v_wmma_f32_16x16x32_bf16 (3 gates x 2 K-chunks = 6/step)
//   - GRU gate update is register-local (accumulator layout == owner lanes)
//   - h kept in registers (hprev[8]); bf16 A-fragment copy double-buffered in
//     LDS -> exactly ONE barrier per timestep
//   - transcendentals via v_exp_f32 + v_rcp_f32 (no IEEE-division sequences)
//   - input-side GEMM folded: gi = base[b] + diff @ (W_ih*vel_W)^T
//   - input streamed through double-buffered LDS (32 steps/chunk)
// ---------------------------------------------------------------------------

typedef __bf16 v16bf __attribute__((ext_vector_type(16)));
typedef float  v8f   __attribute__((ext_vector_type(8)));

#define Bv 4096
#define Tv 2048
#define Hc 64
#define Ec 8
#define Sc 20
#define MB 16
#define NTHREADS 128            // 4 waves
#define CT 32
#define TSTEPS (Tv - 1)         // 2047
#define NCHUNK ((TSTEPS + CT - 1) / CT)
#define INELEMS ((CT + 1) * MB * 2)                  // 1056 floats / chunk
#define RLOADS ((INELEMS + NTHREADS - 1) / NTHREADS) // 9

__device__ __forceinline__ unsigned short f2bf(float f) {
    union { float f; unsigned u; } v; v.f = f;
    unsigned u = v.u;
    u += 0x7FFFu + ((u >> 16) & 1u);
    return (unsigned short)(u >> 16);
}
// v_exp + v_rcp only (no div_scale/div_fixup sequences)
__device__ __forceinline__ float sigmoidf(float x) {
    return __builtin_amdgcn_rcpf(1.0f + __expf(-x));
}
__device__ __forceinline__ float tanh_fast(float x) {
    return 1.0f - 2.0f * __builtin_amdgcn_rcpf(__expf(2.0f * x) + 1.0f);
}

// bf16 A-matrix (16x32) VGPR layout -> linear LDS index within one h-buffer
__device__ __forceinline__ int a_addr(int m, int k) {
    int c  = k >> 5;
    int kk = k & 31;
    int g  = kk >> 3;
    int l  = m + (((g == 1) || (g == 3)) ? 16 : 0);
    int j  = (g == 0) ? kk : (g == 1) ? (kk - 8) : (g == 2) ? (kk - 8) : (kk - 16);
    return (c * 32 + l) * 16 + j;
}

union VU { v16bf v; unsigned short s[16]; };

#define WMMA_BF16(A, B, C) \
    __builtin_amdgcn_wmma_f32_16x16x32_bf16(false, (A), false, (B), (short)0, (C), false, false)

__global__ __launch_bounds__(NTHREADS)
void gru_enc_dec_kernel(const float* __restrict__ input,
                        const float* __restrict__ velW,  const float* __restrict__ velb,
                        const float* __restrict__ encWih, const float* __restrict__ encWhh,
                        const float* __restrict__ encBih, const float* __restrict__ encBhh,
                        const float* __restrict__ decWih, const float* __restrict__ decWhh,
                        const float* __restrict__ decBih, const float* __restrict__ decBhh,
                        const float* __restrict__ linW,  const float* __restrict__ linb,
                        float* __restrict__ out)
{
    __shared__ __align__(32) unsigned short sHbf[2][2 * 32 * 16]; // double-buffered h (bf16, A-frag layout)
    __shared__ __align__(16) float sIN[2][CT + 1][MB][2];         // streamed positions
    __shared__ __align__(16) float sHf[MB][Hc];                   // h f32 (decoder output matvec only)
    __shared__ float sPE[MB][Ec];
    __shared__ __align__(16) float sPX[MB][2];
    __shared__ float sOFF[MB][2];
    __shared__ __align__(16) float sLW[2][Hc];
    __shared__ float sLB[2];

    const int tid   = threadIdx.x;
    const int lane  = tid & 31;
    const int wv    = tid >> 5;              // wave 0..3 -> hidden slice
    const int wgb   = blockIdx.x * MB;
    const int nloc  = lane & 15;
    const int mbase = (lane >> 4) << 3;
    const int kcol  = wv * 16 + nloc;        // this lane's hidden column (0..63)

    // ---- encoder B fragments: gate g tile = Whh rows [64g+16wv, +16), K chunk c ----
    VU be[3][2];
    #pragma unroll
    for (int g = 0; g < 3; ++g)
        #pragma unroll
        for (int c = 0; c < 2; ++c)
            #pragma unroll
            for (int j = 0; j < 16; ++j)
                be[g][c].s[j] = f2bf(encWhh[(64 * g + kcol) * Hc + 32 * c + 8 * (lane >> 4) + (j & 7) + ((j >> 3) << 4)]);

    // ---- per-lane step-invariant scalars (encoder) ----
    float bhE[3], c2e[3][2];
    #pragma unroll
    for (int g = 0; g < 3; ++g) {
        const int col = 64 * g + kcol;
        bhE[g] = encBhh[col];
        float s0 = 0.f, s1 = 0.f;
        #pragma unroll
        for (int e = 0; e < Ec; ++e) {
            float w = encWih[col * Ec + e];
            s0 += w * velW[e * 2 + 0];
            s1 += w * velW[e * 2 + 1];
        }
        c2e[g][0] = s0; c2e[g][1] = s1;
    }

    // ---- cooperative precompute ----
    {   // positional encoding for this block's 16 batch rows (128 entries = 1/thread)
        int m = tid >> 3, e = tid & 7;
        float freq = __expf(-(float)((e >> 1) * 2) * (9.210340371976184f / 8.0f));
        float arg  = (float)(wgb + m) * freq;
        sPE[m][e] = (e & 1) ? __cosf(arg) : __sinf(arg);
    }
    if (tid < 128) sLW[tid >> 6][tid & 63] = linW[tid];
    if (tid < 2)   sLB[tid] = linb[tid];
    if (tid < 32) {
        int m = tid >> 1, cc = tid & 1;
        float vlast = input[((size_t)(wgb + m) * Tv + (Tv - 1)) * 2 + cc];
        float vprev = input[((size_t)(wgb + m) * Tv + (Tv - 2)) * 2 + cc];
        sOFF[m][cc] = vlast;
        sPX[m][cc]  = vlast - vprev;
    }
    #pragma unroll
    for (int i = 0; i < RLOADS; ++i) {       // preload input chunk 0
        int idx = tid + i * NTHREADS;
        if (idx < INELEMS) {
            int ttp = idx >> 5, r = idx & 31, mm = r >> 1, cc = r & 1;
            sIN[0][ttp][mm][cc] = input[((size_t)(wgb + mm) * Tv + ttp) * 2 + cc];
        }
    }
    for (int idx = tid; idx < 2 * 2 * 32 * 16; idx += NTHREADS)
        ((unsigned short*)sHbf)[idx] = 0;    // h0 = 0 in both buffers
    __syncthreads();

    // ---- per-lane gi base: b_ih + (velb + PE[b]) @ Wih^T  (needs sPE) ----
    float baseg[3][8];
    #pragma unroll
    for (int g = 0; g < 3; ++g) {
        const int col = 64 * g + kcol;
        #pragma unroll
        for (int d = 0; d < 8; ++d) {
            const int m = mbase + d;
            float s = encBih[col];
            #pragma unroll
            for (int e = 0; e < Ec; ++e)
                s += (velb[e] + sPE[m][e]) * encWih[col * Ec + e];
            baseg[g][d] = s;
        }
    }

    // step-invariant bf16-h store addresses (within one buffer)
    int saddr[8];
    #pragma unroll
    for (int d = 0; d < 8; ++d) saddr[d] = a_addr(mbase + d, kcol);

    float hprev[8] = {0.f, 0.f, 0.f, 0.f, 0.f, 0.f, 0.f, 0.f};
    int par = 0;

    // ================= encoder: 2047 sequential steps =================
    for (int chk = 0; chk < NCHUNK; ++chk) {
        const int p = chk & 1;
        float rbuf[RLOADS];
        const bool havenext = (chk + 1 < NCHUNK);
        if (havenext) {                       // issue next-chunk global loads early
            #pragma unroll
            for (int i = 0; i < RLOADS; ++i) {
                int idx = tid + i * NTHREADS;
                if (idx < INELEMS) {
                    int ttp = idx >> 5, r = idx & 31, mm = r >> 1, cc = r & 1;
                    int tg = (chk + 1) * CT + ttp; if (tg > Tv - 1) tg = Tv - 1;
                    rbuf[i] = input[((size_t)(wgb + mm) * Tv + tg) * 2 + cc];
                }
            }
        }
        const int t0 = chk * CT;
        #pragma unroll 1
        for (int tt = 0; tt < CT; ++tt) {
            if (t0 + tt < TSTEPS) {           // uniform branch
                const unsigned short* hb = sHbf[par];
                v16bf a0 = *(const v16bf*)&hb[lane * 16];
                v16bf a1 = *(const v16bf*)&hb[(32 + lane) * 16];
                v8f aR = { bhE[0],bhE[0],bhE[0],bhE[0],bhE[0],bhE[0],bhE[0],bhE[0] };
                v8f aZ = { bhE[1],bhE[1],bhE[1],bhE[1],bhE[1],bhE[1],bhE[1],bhE[1] };
                v8f aN = { bhE[2],bhE[2],bhE[2],bhE[2],bhE[2],bhE[2],bhE[2],bhE[2] };
                aR = WMMA_BF16(a0, be[0][0].v, aR); aR = WMMA_BF16(a1, be[0][1].v, aR);
                aZ = WMMA_BF16(a0, be[1][0].v, aZ); aZ = WMMA_BF16(a1, be[1][1].v, aZ);
                aN = WMMA_BF16(a0, be[2][0].v, aN); aN = WMMA_BF16(a1, be[2][1].v, aN);

                float rA[16], rB[16];
                const float* pA = &sIN[p][tt][mbase][0];
                const float* pB = &sIN[p][tt + 1][mbase][0];
                #pragma unroll
                for (int i = 0; i < 16; ++i) { rA[i] = pA[i]; rB[i] = pB[i]; }

                unsigned short* hw = sHbf[par ^ 1];
                #pragma unroll
                for (int d = 0; d < 8; ++d) {
                    float dx = rB[2 * d]     - rA[2 * d];
                    float dy = rB[2 * d + 1] - rA[2 * d + 1];
                    float r = sigmoidf(baseg[0][d] + dx * c2e[0][0] + dy * c2e[0][1] + aR[d]);
                    float z = sigmoidf(baseg[1][d] + dx * c2e[1][0] + dy * c2e[1][1] + aZ[d]);
                    float n = tanh_fast(baseg[2][d] + dx * c2e[2][0] + dy * c2e[2][1] + r * aN[d]);
                    hprev[d] = n + z * (hprev[d] - n);
                    hw[saddr[d]] = f2bf(hprev[d]);
                }
                par ^= 1;
                __syncthreads();              // single barrier per step
            }
        }
        if (havenext) {
            #pragma unroll
            for (int i = 0; i < RLOADS; ++i) {
                int idx = tid + i * NTHREADS;
                if (idx < INELEMS) {
                    int ttp = idx >> 5, r = idx & 31, mm = r >> 1, cc = r & 1;
                    sIN[1 - p][ttp][mm][cc] = rbuf[i];
                }
            }
        }
        __syncthreads();
    }

    // ================= decoder: 20 autoregressive steps =================
    VU bd[3][2];
    #pragma unroll
    for (int g = 0; g < 3; ++g)
        #pragma unroll
        for (int c = 0; c < 2; ++c)
            #pragma unroll
            for (int j = 0; j < 16; ++j)
                bd[g][c].s[j] = f2bf(decWhh[(64 * g + kcol) * Hc + 32 * c + 8 * (lane >> 4) + (j & 7) + ((j >> 3) << 4)]);

    float bhD[3], c2d[3][2], dbase[3];
    #pragma unroll
    for (int g = 0; g < 3; ++g) {
        const int col = 64 * g + kcol;
        bhD[g] = decBhh[col];
        float s0 = 0.f, s1 = 0.f, sb = decBih[col];
        #pragma unroll
        for (int e = 0; e < Ec; ++e) {
            float w = decWih[col * Ec + e];
            s0 += w * velW[e * 2 + 0];
            s1 += w * velW[e * 2 + 1];
            sb += w * velb[e];
        }
        c2d[g][0] = s0; c2d[g][1] = s1; dbase[g] = sb;
    }

    #pragma unroll 1
    for (int s = 0; s < Sc; ++s) {
        const unsigned short* hb = sHbf[par];
        v16bf a0 = *(const v16bf*)&hb[lane * 16];
        v16bf a1 = *(const v16bf*)&hb[(32 + lane) * 16];
        v8f aR = { bhD[0],bhD[0],bhD[0],bhD[0],bhD[0],bhD[0],bhD[0],bhD[0] };
        v8f aZ = { bhD[1],bhD[1],bhD[1],bhD[1],bhD[1],bhD[1],bhD[1],bhD[1] };
        v8f aN = { bhD[2],bhD[2],bhD[2],bhD[2],bhD[2],bhD[2],bhD[2],bhD[2] };
        aR = WMMA_BF16(a0, bd[0][0].v, aR); aR = WMMA_BF16(a1, bd[0][1].v, aR);
        aZ = WMMA_BF16(a0, bd[1][0].v, aZ); aZ = WMMA_BF16(a1, bd[1][1].v, aZ);
        aN = WMMA_BF16(a0, bd[2][0].v, aN); aN = WMMA_BF16(a1, bd[2][1].v, aN);

        float qp[16];
        const float* pq = &sPX[mbase][0];
        #pragma unroll
        for (int i = 0; i < 16; ++i) qp[i] = pq[i];

        unsigned short* hw = sHbf[par ^ 1];
        #pragma unroll
        for (int d = 0; d < 8; ++d) {
            float dx = qp[2 * d], dy = qp[2 * d + 1];
            float r = sigmoidf(dbase[0] + dx * c2d[0][0] + dy * c2d[0][1] + aR[d]);
            float z = sigmoidf(dbase[1] + dx * c2d[1][0] + dy * c2d[1][1] + aZ[d]);
            float n = tanh_fast(dbase[2] + dx * c2d[2][0] + dy * c2d[2][1] + r * aN[d]);
            hprev[d] = n + z * (hprev[d] - n);
            hw[saddr[d]] = f2bf(hprev[d]);
            sHf[mbase + d][kcol] = hprev[d];
        }
        par ^= 1;
        __syncthreads();
        // x = h @ linW^T + linb + prev_x ; emit x + offset ; prev_x <- x
        if (tid < 32) {
            int m = tid >> 1, j = tid & 1;
            float xv = sLB[j] + sPX[m][j];
            #pragma unroll
            for (int k = 0; k < Hc; ++k) xv += sHf[m][k] * sLW[j][k];
            out[(((size_t)(wgb + m)) * Sc + s) * 2 + j] = xv + sOFF[m][j];
            sPX[m][j] = xv;
        }
        __syncthreads();
    }
}

extern "C" void kernel_launch(void* const* d_in, const int* in_sizes, int n_in,
                              void* d_out, int out_size, void* d_ws, size_t ws_size,
                              hipStream_t stream) {
    (void)in_sizes; (void)n_in; (void)out_size; (void)d_ws; (void)ws_size;
    const float* input  = (const float*)d_in[0];
    const float* velW   = (const float*)d_in[1];
    const float* velb   = (const float*)d_in[2];
    const float* encWih = (const float*)d_in[3];
    const float* encWhh = (const float*)d_in[4];
    const float* encBih = (const float*)d_in[5];
    const float* encBhh = (const float*)d_in[6];
    const float* decWih = (const float*)d_in[7];
    const float* decWhh = (const float*)d_in[8];
    const float* decBih = (const float*)d_in[9];
    const float* decBhh = (const float*)d_in[10];
    const float* linW   = (const float*)d_in[11];
    const float* linb   = (const float*)d_in[12];
    float* out = (float*)d_out;

    dim3 grid(Bv / MB);       // 256 blocks
    dim3 block(NTHREADS);     // 128 threads = 4 wave32 waves
    gru_enc_dec_kernel<<<grid, block, 0, stream>>>(
        input, velW, velb, encWih, encWhh, encBih, encBhh,
        decWih, decWhh, decBih, decBhh, linW, linb, out);
}
